// MyRNN_73323681677970
// MI455X (gfx1250) — matrix operand order repaired and checked by reading, not compiled, not run on hardware
//
#include <hip/hip_runtime.h>

// ---------------------------------------------------------------------------
// RNN forward for MI455X (gfx1250, wave32, WMMA + TDM).
//   inputs [64][2048][256] f32, Wxh[256][512], Whh[512][512], Why[512][256],
//   bh[512], by[256]  ->  out [64][256] f32
//
// Phase 1 (whole GPU): xproj[t][b][h] = bf16( X@Wxh + bh )      (WMMA bf16)
// Phase 2 (one WGP, persistent): H_t = tanh(xproj_t + H@Whh)    (WMMA bf16,
//          Whh resident in VGPRs in B-operand layout, H in LDS,
//          xproj slices double-buffered into LDS by the Tensor Data Mover)
// Phase 3: out = H_final @ Why + by                              (WMMA bf16)
// ---------------------------------------------------------------------------

typedef __attribute__((ext_vector_type(16))) __bf16        v16bf;
typedef __attribute__((ext_vector_type(8)))  float         v8f;
typedef __attribute__((ext_vector_type(4))) unsigned int   u32x4;
typedef __attribute__((ext_vector_type(8)))  int           i32x8;
typedef __attribute__((ext_vector_type(4)))  int           i32x4;

union TileU { v16bf v; u32x4 q[2]; };

typedef unsigned short ushort_t;

#define RNN_I 256
#define RNN_H 512
#define RNN_O 256
#define RNN_B 64
#define RNN_T 2048

__device__ __forceinline__ ushort_t f2bf(float f) {
    union { float f; unsigned int u; } x; x.f = f;
    unsigned int u = x.u;
    unsigned int r = (u + 0x7FFFu + ((u >> 16) & 1u)) >> 16;   // RNE
    return (ushort_t)r;
}
__device__ __forceinline__ float bf2f(ushort_t h) {
    union { unsigned int u; float f; } x; x.u = ((unsigned int)h) << 16;
    return x.f;
}

// ---------------------------------------------------------------------------
// TDM: one-shot DMA of a contiguous 64 KB bf16 slice (64x512) into LDS.
// D# per CDNA5 ISA ch.8: group0 = {count|flags, lds_addr, global_addr lo,
// global_addr hi | type=2}; group1 = {mask/data_size, dims, tile dims, strides}.
// Tile described as one row of 32768 2-byte elements.
// ---------------------------------------------------------------------------
__device__ __forceinline__ void tdm_load_xp_slice(const ushort_t* gsrc,
                                                  unsigned int ldsByteOff) {
    unsigned long long ga = (unsigned long long)(size_t)gsrc;
    u32x4 g0;
    g0.x = 1u;                                          // count=1, user mode
    g0.y = ldsByteOff;                                  // LDS byte address
    g0.z = (unsigned int)(ga & 0xFFFFFFFFu);            // global_addr[31:0]
    g0.w = (unsigned int)((ga >> 32) & 0x01FFFFFFu)     // global_addr[56:32]
         | (2u << 30);                                  // type = 2 ("image")
    i32x8 g1;
    g1[0] = 0x00010000;                                 // data_size=1 -> 2 bytes
    g1[1] = (int)0x80000000;                            // tensor_dim0 = 32768 (bits 79:48, low half)
    g1[2] = 0x00010000;                                 // tensor_dim1 = 1     (bits 111:80)
    g1[3] = (int)0x80000000;                            // tile_dim0   = 32768 (bits 127:112)
    g1[4] = 1;                                          // tile_dim1 = 1, tile_dim2 = 0
    g1[5] = 32768;                                      // tensor_dim0_stride
    g1[6] = 0;
    g1[7] = 0;
    i32x4 g2 = {0, 0, 0, 0};                            // <=2D tensor: unused
    i32x4 g3 = {0, 0, 0, 0};
#if defined(__clang_major__) && (__clang_major__ >= 23)
    i32x8 g4 = {0, 0, 0, 0, 0, 0, 0, 0};
    __builtin_amdgcn_tensor_load_to_lds(g0, g1, g2, g3, g4, 0);
#else
    __builtin_amdgcn_tensor_load_to_lds(g0, g1, g2, g3, 0);
#endif
}

// ---------------------------------------------------------------------------
// Pack a K x N f32 weight into WMMA B-operand order (bf16):
//   record[((nb*KC + kc)*32 + lane)*16 + j] = W[kc*32 + 8*(lane/16) + (j%8) + 16*(j/8)][nb*16 + lane%16]
// so a wave loads one 32x16 B tile as two contiguous b128s per lane.
// ---------------------------------------------------------------------------
__global__ void pack_w_kernel(const float* __restrict__ W, int K, int N,
                              ushort_t* __restrict__ out) {
    int tid = blockIdx.x * blockDim.x + threadIdx.x;
    int KC = K >> 5;
    int NB = N >> 4;
    int total = NB * KC * 32;
    if (tid >= total) return;
    int lane  = tid & 31;
    int kc    = (tid >> 5) % KC;
    int nb    = tid / (32 * KC);
    int kbase = (kc << 5) + ((lane >> 4) << 3);
    int n     = (nb << 4) + (lane & 15);
    ushort_t* dst = out + ((size_t)tid << 4);
#pragma unroll
    for (int j = 0; j < 16; ++j)
        dst[j] = f2bf(W[(size_t)(kbase + (j & 7) + ((j >> 3) << 4)) * N + n]);
}

// ---------------------------------------------------------------------------
// Phase 1: xproj[t][b][h] = bf16( sum_i X[b][t][i] * Wxh[i][h] + bh[h] )
// One block per 16-row M strip (fixed t, 16 consecutive b). 16 waves, each
// owns two 16-column N tiles. A strip staged in LDS as bf16.
// ---------------------------------------------------------------------------
__global__ __launch_bounds__(512, 1)
void xproj_kernel(const float* __restrict__ X,
                  const ushort_t* __restrict__ WxhP,
                  const float* __restrict__ bh,
                  ushort_t* __restrict__ xp) {
    __shared__ ushort_t Alds[16][264];                 // 16 x 256, +8 pad (bank-safe)

    const int t  = blockIdx.x >> 2;
    const int b0 = (blockIdx.x & 3) << 4;
    const int tid = threadIdx.x;

    {   // cooperative stage: 512 threads x 8 f32 each = 16x256
        int i  = tid >> 5;
        int kk = (tid & 31) << 3;
        const float* src = X + ((size_t)(b0 + i) * RNN_T + t) * RNN_I + kk;
#pragma unroll
        for (int j = 0; j < 8; ++j) Alds[i][kk + j] = f2bf(src[j]);
    }
    __syncthreads();

    const int wave = tid >> 5, lane = tid & 31;
    const int lhi = lane >> 4, llo = lane & 15;
    const int nb0 = wave << 1;

    float bias0 = bh[(nb0 << 4) + llo];
    float bias1 = bh[((nb0 + 1) << 4) + llo];
    v8f acc0, acc1;
#pragma unroll
    for (int v = 0; v < 8; ++v) { acc0[v] = bias0; acc1[v] = bias1; }

#pragma unroll
    for (int kc = 0; kc < (RNN_I / 32); ++kc) {
        TileU a;
        const ushort_t* ap = &Alds[llo][(kc << 5) + (lhi << 3)];
        a.q[0] = *(const u32x4*)(ap);
        a.q[1] = *(const u32x4*)(ap + 16);
        TileU b0t, b1t;
        const u32x4* bp0 = (const u32x4*)(WxhP + ((((size_t)nb0) * (RNN_I / 32) + kc) * 32 + lane) * 16);
        const u32x4* bp1 = (const u32x4*)(WxhP + ((((size_t)nb0 + 1) * (RNN_I / 32) + kc) * 32 + lane) * 16);
        b0t.q[0] = bp0[0]; b0t.q[1] = bp0[1];
        b1t.q[0] = bp1[0]; b1t.q[1] = bp1[1];
        acc0 = __builtin_amdgcn_wmma_f32_16x16x32_bf16(false, a.v, false, b0t.v,
                                                       (short)0, acc0, false, false);
        acc1 = __builtin_amdgcn_wmma_f32_16x16x32_bf16(false, a.v, false, b1t.v,
                                                       (short)0, acc1, false, false);
    }

    // store: element v <-> row (b0 + v + 8*lhi), col nb*16 + llo
#pragma unroll
    for (int v = 0; v < 8; ++v) {
        int b = b0 + v + (lhi << 3);
        size_t base = ((size_t)t * RNN_B + b) * RNN_H;
        xp[base + (nb0 << 4) + llo]       = f2bf(acc0[v]);
        xp[base + ((nb0 + 1) << 4) + llo] = f2bf(acc1[v]);
    }
}

// ---------------------------------------------------------------------------
// Phase 2: persistent single workgroup. 16 waves; wave w owns columns
// [32w, 32w+32) of H. Whh B tiles (512 KB bf16) live in VGPRs (256/wave).
// LDS layout (dynamic, 197632 B total):
//   [0      , 66560)  H  (64 x 520 bf16, +8 pad rows -> conflict-free b128)
//   [66560  , 132096) xp slice buffer 0  (64 x 512 bf16)
//   [132096 , 197632) xp slice buffer 1
// TDM prefetches slice t+1 while WMMAs run on slice t's recurrence.
// ---------------------------------------------------------------------------
#define HS 520
#define HLDS_BYTES   (RNN_B * HS * 2)            /* 66560  */
#define XBUF_BYTES   (RNN_B * RNN_H * 2)         /* 65536  */
#define SMEM_BYTES   (HLDS_BYTES + 2 * XBUF_BYTES)

__global__ __launch_bounds__(512, 1)
void rnn_scan_kernel(const ushort_t* __restrict__ xp,
                     const ushort_t* __restrict__ WhhP,
                     ushort_t* __restrict__ Hfin) {
    extern __shared__ ushort_t smem[];
    ushort_t* Hlds = smem;                                    // [64][HS]

    const int tid = threadIdx.x;
    const int wave = tid >> 5, lane = tid & 31;
    const int lhi = lane >> 4, llo = lane & 15;

    // kick off TDM for slice 0 as early as possible
    if (wave == 0) tdm_load_xp_slice(xp, HLDS_BYTES);

    // H0 = 0
    for (int i = tid; i < RNN_B * HS; i += 512) Hlds[i] = 0;

    // Load resident Whh B tiles: 2 n-blocks x 16 k-chunks, 32 B each per lane.
    v16bf Bt[2][16];
#pragma unroll
    for (int q = 0; q < 2; ++q) {
        int nb = (wave << 1) + q;
#pragma unroll
        for (int kc = 0; kc < 16; ++kc) {
            TileU u;
            const u32x4* p = (const u32x4*)(WhhP + (((size_t)nb * 16 + kc) * 32 + lane) * 16);
            u.q[0] = p[0]; u.q[1] = p[1];
            Bt[q][kc] = u.v;
        }
    }
    __syncthreads();

    for (int t = 0; t < RNN_T; ++t) {
        // prefetch next xproj slice via the Tensor Data Mover (wave 0 only).
        // End-of-step barrier of step t-1 guarantees nobody still reads the
        // parity-(t+1) buffer when this overwrite starts.
        if (wave == 0 && (t + 1) < RNN_T) {
            tdm_load_xp_slice(xp + (size_t)(t + 1) * RNN_B * RNN_H,
                              HLDS_BYTES + (unsigned int)((t + 1) & 1) * XBUF_BYTES);
        }

        v8f acc[4][2];
#pragma unroll
        for (int m = 0; m < 4; ++m)
#pragma unroll
            for (int q = 0; q < 2; ++q)
#pragma unroll
                for (int v = 0; v < 8; ++v) acc[m][q][v] = 0.0f;

#pragma unroll
        for (int kc = 0; kc < 16; ++kc) {
#pragma unroll
            for (int m = 0; m < 4; ++m) {
                TileU a;
                const ushort_t* ap = &Hlds[(size_t)((m << 4) + llo) * HS + (kc << 5) + (lhi << 3)];
                a.q[0] = *(const u32x4*)(ap);
                a.q[1] = *(const u32x4*)(ap + 16);
                acc[m][0] = __builtin_amdgcn_wmma_f32_16x16x32_bf16(false, a.v, false, Bt[0][kc],
                                                                   (short)0, acc[m][0], false, false);
                acc[m][1] = __builtin_amdgcn_wmma_f32_16x16x32_bf16(false, a.v, false, Bt[1][kc],
                                                                   (short)0, acc[m][1], false, false);
            }
        }

        // TDM completes in order: <=1 outstanding means slice t has landed.
        if (wave == 0) {
            if (t == RNN_T - 1) __builtin_amdgcn_s_wait_tensorcnt(0);
            else                __builtin_amdgcn_s_wait_tensorcnt(1);
        }
        __syncthreads();   // A-tile reads of H_t done + slice t visible to all

        // active xp slice buffer (computed, not a pointer table, to avoid
        // constant-folded LDS addrspacecast initializers)
        const ushort_t* xrow = smem + (HLDS_BYTES / 2)
                             + (size_t)(t & 1) * (XBUF_BYTES / 2);
#pragma unroll
        for (int m = 0; m < 4; ++m) {
#pragma unroll
            for (int q = 0; q < 2; ++q) {
                int n = (((wave << 1) + q) << 4) + llo;
#pragma unroll
                for (int v = 0; v < 8; ++v) {
                    int mm = (m << 4) + v + (lhi << 3);
                    float s = acc[m][q][v] + bf2f(xrow[(size_t)mm * RNN_H + n]);
                    Hlds[(size_t)mm * HS + n] = f2bf(tanhf(s));
                }
            }
        }
        __syncthreads();   // H_{t+1} fully written before next step reads
    }

    // spill final H to global (bf16 row-major [64][512])
    for (int i = tid; i < RNN_B * RNN_H; i += 512) {
        int r = i >> 9, c = i & 511;
        Hfin[i] = Hlds[(size_t)r * HS + c];
    }
}

// ---------------------------------------------------------------------------
// Phase 3: out[64][256] = Hfin @ Why + by. One block, 16 waves; wave w owns
// 16-column N tile w; 4 M tiles each; K = 512 (16 chunks).
// ---------------------------------------------------------------------------
__global__ __launch_bounds__(512, 1)
void out_kernel(const ushort_t* __restrict__ Hfin,
                const ushort_t* __restrict__ WhyP,
                const float* __restrict__ by,
                float* __restrict__ out) {
    const int tid = threadIdx.x;
    const int wave = tid >> 5, lane = tid & 31;
    const int lhi = lane >> 4, llo = lane & 15;

    v8f acc[4];
#pragma unroll
    for (int m = 0; m < 4; ++m)
#pragma unroll
        for (int v = 0; v < 8; ++v) acc[m][v] = 0.0f;

#pragma unroll
    for (int kc = 0; kc < 16; ++kc) {
        TileU b;
        const u32x4* bp = (const u32x4*)(WhyP + (((size_t)wave * 16 + kc) * 32 + lane) * 16);
        b.q[0] = bp[0]; b.q[1] = bp[1];
#pragma unroll
        for (int m = 0; m < 4; ++m) {
            TileU a;
            const ushort_t* ap = Hfin + (size_t)((m << 4) + llo) * RNN_H + (kc << 5) + (lhi << 3);
            a.q[0] = *(const u32x4*)(ap);
            a.q[1] = *(const u32x4*)(ap + 16);
            acc[m] = __builtin_amdgcn_wmma_f32_16x16x32_bf16(false, a.v, false, b.v,
                                                             (short)0, acc[m], false, false);
        }
    }

    int n = (wave << 4) + llo;
    float bias = by[n];
#pragma unroll
    for (int m = 0; m < 4; ++m)
#pragma unroll
        for (int v = 0; v < 8; ++v) {
            int r = (m << 4) + v + (lhi << 3);
            out[(size_t)r * RNN_O + n] = acc[m][v] + bias;
        }
}

// ---------------------------------------------------------------------------
// Launch
// ---------------------------------------------------------------------------
extern "C" void kernel_launch(void* const* d_in, const int* in_sizes, int n_in,
                              void* d_out, int out_size, void* d_ws, size_t ws_size,
                              hipStream_t stream) {
    (void)in_sizes; (void)n_in; (void)out_size; (void)ws_size;

    const float* X   = (const float*)d_in[0];   // [64][2048][256]
    const float* Wxh = (const float*)d_in[1];   // [256][512]
    const float* Whh = (const float*)d_in[2];   // [512][512]
    const float* Why = (const float*)d_in[3];   // [512][256]
    const float* bh  = (const float*)d_in[4];   // [512]
    const float* by  = (const float*)d_in[5];   // [256]
    float* out = (float*)d_out;                 // [64][256]

    char* ws = (char*)d_ws;
    // workspace layout (bytes)
    ushort_t* WxhP = (ushort_t*)(ws + 0);                    // 32*8*32*16*2   = 256 KB
    ushort_t* WhhP = (ushort_t*)(ws + 262144);               // 32*16*32*16*2  = 512 KB
    ushort_t* WhyP = (ushort_t*)(ws + 262144 + 524288);      // 16*16*32*16*2  = 256 KB
    ushort_t* Hfin = (ushort_t*)(ws + 1048576);              // 64*512*2       = 64 KB
    ushort_t* xpW  = (ushort_t*)(ws + 1114112);              // 2048*64*512*2  = 128 MB

    // pack weights into WMMA B-operand bf16 layout
    pack_w_kernel<<<(8192 + 255) / 256, 256, 0, stream>>>(Wxh, RNN_I, RNN_H, WxhP);
    pack_w_kernel<<<(16384 + 255) / 256, 256, 0, stream>>>(Whh, RNN_H, RNN_H, WhhP);
    pack_w_kernel<<<(8192 + 255) / 256, 256, 0, stream>>>(Why, RNN_H, RNN_O, WhyP);

    // phase 1: big parallel GEMM, 131072 rows / 16 per block
    xproj_kernel<<<(RNN_T * RNN_B) / 16, 512, 0, stream>>>(X, WxhP, bh, xpW);

    // phase 2: serial scan, one persistent workgroup, 193 KB dynamic LDS
    rnn_scan_kernel<<<1, 512, SMEM_BYTES, stream>>>(xpW, WhhP, Hfin);

    // phase 3: output GEMM
    out_kernel<<<1, 512, 0, stream>>>(Hfin, WhyP, by, out);
}
